// QA_CNN_53687091200212
// MI455X (gfx1250) — matrix operand order repaired and checked by reading, not compiled, not run on hardware
//
#include <hip/hip_runtime.h>

// ---------------------------------------------------------------------------
// QA-CNN: per-position linear map over a ctx=5 sliding window == batched GEMM
//   A_b [L x 1500] (overlapping contiguous rows of padded x)  @  W [1500 x 512]
// fp32 emulated via 3x bf16 WMMA (hi*Whi + hi*Wlo + lo*Whi), f32 accumulate.
// Block = (batch, 32 L-rows) x full F=512; 8 waves, each wave: 2 M-tiles x
// 4 N-subtiles sharing B fragments -> 24 WMMA per K-step per wave.
// ---------------------------------------------------------------------------

typedef __attribute__((ext_vector_type(16))) __bf16 v16bf;
typedef __attribute__((ext_vector_type(8)))  __bf16 v8bf;
typedef __attribute__((ext_vector_type(8)))  float  v8f;
typedef __attribute__((ext_vector_type(4)))  float  v4f;

#define EDIM 300
#define FDIM 512
#define KDIM 1500
#define NKB  47          // ceil(1500/32) K-blocks of 32 (tail zero-padded)
#define WP_ELEMS (NKB * 2 * FDIM * 16)   // [kb][half][n][16] = 770048

union AFrag { v16bf v; v8bf h[2]; };

// --- pack W into WMMA B-fragment order, split into bf16 hi/lo -------------
// Wp[kb][g][n][i] = W[kb*32 + (i < 8 ? g*8 + i : 16 + g*8 + i - 8)][n]   (zero for k >= 1500)
// A lane (column n, half g) reads its 16 K-values as one contiguous 32B.
__global__ void qa_pack_w(const float* __restrict__ W,
                          __bf16* __restrict__ wp_hi, __bf16* __restrict__ wp_lo) {
    int idx = blockIdx.x * blockDim.x + threadIdx.x;   // < WP_ELEMS
    int i  = idx & 15;
    int n  = (idx >> 4) & 511;
    int g  = (idx >> 13) & 1;
    int kb = idx >> 14;
    int k  = kb * 32 + (i < 8 ? g * 8 + i : 16 + g * 8 + (i - 8));
    float v = (k < KDIM) ? W[(size_t)k * FDIM + n] : 0.0f;
    __bf16 hi = (__bf16)v;
    __bf16 lo = (__bf16)(v - (float)hi);
    wp_hi[idx] = hi;
    wp_lo[idx] = lo;
}

// --- build zero-padded bf16 hi/lo copy of x -------------------------------
// xpad layout per batch: (L+4) rows of 300 + 32 zero tail (covers K pad 1504).
// z-row l == xpad[l*300 .. l*300+1499]  (contiguous!)
__global__ void qa_pack_x(const float* __restrict__ x,
                          __bf16* __restrict__ hi, __bf16* __restrict__ lo,
                          int L, int S) {
    int r = blockIdx.x * blockDim.x + threadIdx.x;
    int b = blockIdx.y;
    if (r >= S) return;
    float v = 0.0f;
    int inner = (L + 4) * EDIM;
    if (r < inner) {
        int row  = r / EDIM;
        int e    = r - row * EDIM;
        int irow = row - 2;                  // ctx//2 = 2
        if (irow >= 0 && irow < L) v = x[((size_t)b * L + irow) * EDIM + e];
    }
    __bf16 h = (__bf16)v;
    size_t o = (size_t)b * S + r;
    hi[o] = h;
    lo[o] = (__bf16)(v - (float)h);
}

// --- main GEMM ------------------------------------------------------------
__global__ __launch_bounds__(256) void qa_gemm(
    const __bf16* __restrict__ xp_hi, const __bf16* __restrict__ xp_lo,
    const __bf16* __restrict__ wp_hi, const __bf16* __restrict__ wp_lo,
    const float* __restrict__ bias, float* __restrict__ out,
    int L, int S) {

    const int lane = threadIdx.x & 31;
    const int wave = threadIdx.x >> 5;
    const int hh   = lane >> 4;      // half: A -> K{0..7,16..23}; B -> K{0..7,16..23}
    const int mn   = lane & 15;      // row (A) / column (B/C/D) index
    const int l0   = blockIdx.x * 32;
    const int bb   = blockIdx.y;

    // A: rows (l0+mn) and (l0+16+mn) of the implicit z-matrix = contiguous
    // spans of the padded input.
    const size_t arow0 = (size_t)bb * S + (size_t)(l0 + mn) * EDIM + hh * 8;
    const size_t arow1 = arow0 + (size_t)16 * EDIM;
    const __bf16* pa_hi0 = xp_hi + arow0;
    const __bf16* pa_lo0 = xp_lo + arow0;
    const __bf16* pa_hi1 = xp_hi + arow1;
    const __bf16* pa_lo1 = xp_lo + arow1;

    // B: packed fragments; per K-block stride 16384, per subtile stride 256
    const size_t boff = (size_t)hh * 8192 + (size_t)(wave * 64 + mn) * 16;
    const __bf16* pb_hi = wp_hi + boff;
    const __bf16* pb_lo = wp_lo + boff;

    v8f acc0[4] = {v8f{}, v8f{}, v8f{}, v8f{}};
    v8f acc1[4] = {v8f{}, v8f{}, v8f{}, v8f{}};

    for (int kb = 0; kb < NKB; ++kb) {
        const int ko = kb * 32;
        AFrag a0h, a0l, a1h, a1l;
        a0h.h[0] = *(const v8bf*)(pa_hi0 + ko);
        a0h.h[1] = *(const v8bf*)(pa_hi0 + ko + 16);
        a0l.h[0] = *(const v8bf*)(pa_lo0 + ko);
        a0l.h[1] = *(const v8bf*)(pa_lo0 + ko + 16);
        a1h.h[0] = *(const v8bf*)(pa_hi1 + ko);
        a1h.h[1] = *(const v8bf*)(pa_hi1 + ko + 16);
        a1l.h[0] = *(const v8bf*)(pa_lo1 + ko);
        a1l.h[1] = *(const v8bf*)(pa_lo1 + ko + 16);

        const __bf16* bh = pb_hi + (size_t)kb * 16384;
        const __bf16* bl = pb_lo + (size_t)kb * 16384;
        v16bf wh[4], wl[4];
#pragma unroll
        for (int t = 0; t < 4; ++t) {
            wh[t] = *(const v16bf*)(bh + t * 256);   // 32B contiguous
            wl[t] = *(const v16bf*)(bl + t * 256);
        }
#pragma unroll
        for (int t = 0; t < 4; ++t) {
            acc0[t] = __builtin_amdgcn_wmma_f32_16x16x32_bf16(
                false, a0h.v, false, wh[t], (short)0, acc0[t], false, false);
            acc1[t] = __builtin_amdgcn_wmma_f32_16x16x32_bf16(
                false, a1h.v, false, wh[t], (short)0, acc1[t], false, false);
            acc0[t] = __builtin_amdgcn_wmma_f32_16x16x32_bf16(
                false, a0h.v, false, wl[t], (short)0, acc0[t], false, false);
            acc1[t] = __builtin_amdgcn_wmma_f32_16x16x32_bf16(
                false, a1h.v, false, wl[t], (short)0, acc1[t], false, false);
            acc0[t] = __builtin_amdgcn_wmma_f32_16x16x32_bf16(
                false, a0l.v, false, wh[t], (short)0, acc0[t], false, false);
            acc1[t] = __builtin_amdgcn_wmma_f32_16x16x32_bf16(
                false, a1l.v, false, wh[t], (short)0, acc1[t], false, false);
        }
    }

    // Epilogue: C fragment lane mn holds column f; VGPR r holds row m =
    // hh*8 + r. Output [B, F, L]: l contiguous -> b128 stores.
#pragma unroll
    for (int t = 0; t < 4; ++t) {
        int f = wave * 64 + t * 16 + mn;
        float bv = bias[f];
        size_t ob = ((size_t)bb * FDIM + f) * L + l0 + hh * 8;
        v8f c = acc0[t];
        v4f s0 = {c[0] + bv, c[1] + bv, c[2] + bv, c[3] + bv};
        v4f s1 = {c[4] + bv, c[5] + bv, c[6] + bv, c[7] + bv};
        *(v4f*)(out + ob)     = s0;
        *(v4f*)(out + ob + 4) = s1;
        v8f d = acc1[t];
        v4f s2 = {d[0] + bv, d[1] + bv, d[2] + bv, d[3] + bv};
        v4f s3 = {d[4] + bv, d[5] + bv, d[6] + bv, d[7] + bv};
        *(v4f*)(out + ob + 16) = s2;
        *(v4f*)(out + ob + 20) = s3;
    }
}

extern "C" void kernel_launch(void* const* d_in, const int* in_sizes, int n_in,
                              void* d_out, int out_size, void* d_ws, size_t ws_size,
                              hipStream_t stream) {
    const float* q    = (const float*)d_in[0];   // [256, 64, 300]
    const float* a    = (const float*)d_in[1];   // [256, 256, 300]
    const float* W    = (const float*)d_in[2];   // [1500, 512]
    const float* bias = (const float*)d_in[3];   // [512]
    float* out = (float*)d_out;

    const int B = 256, LQ = 64, LA = 256;
    const int SQ = (LQ + 4) * EDIM + 32;   // 20432 (tail covers K-pad reads)
    const int SA = (LA + 4) * EDIM + 32;   // 78032

    char* ws = (char*)d_ws;
    size_t off = 0;
    auto carve = [&](size_t bytes) {
        void* p = ws + off;
        off += (bytes + 255) & ~(size_t)255;
        return p;
    };
    __bf16* wp_hi = (__bf16*)carve((size_t)WP_ELEMS * 2);
    __bf16* wp_lo = (__bf16*)carve((size_t)WP_ELEMS * 2);
    __bf16* q_hi  = (__bf16*)carve((size_t)B * SQ * 2);
    __bf16* q_lo  = (__bf16*)carve((size_t)B * SQ * 2);
    __bf16* a_hi  = (__bf16*)carve((size_t)B * SA * 2);
    __bf16* a_lo  = (__bf16*)carve((size_t)B * SA * 2);
    (void)ws_size; (void)in_sizes; (void)n_in; (void)out_size;

    qa_pack_w<<<WP_ELEMS / 256, 256, 0, stream>>>(W, wp_hi, wp_lo);
    qa_pack_x<<<dim3((SQ + 255) / 256, B), 256, 0, stream>>>(q, q_hi, q_lo, LQ, SQ);
    qa_pack_x<<<dim3((SA + 255) / 256, B), 256, 0, stream>>>(a, a_hi, a_lo, LA, SA);

    qa_gemm<<<dim3(LQ / 32, B), 256, 0, stream>>>(q_hi, q_lo, wp_hi, wp_lo, bias,
                                                  out, LQ, SQ);
    qa_gemm<<<dim3(LA / 32, B), 256, 0, stream>>>(a_hi, a_lo, wp_hi, wp_lo, bias,
                                                  out + (size_t)B * FDIM * LQ, LA, SA);
}
